// DeCorrelatedGroupNorm_48172353192268
// MI455X (gfx1250) — matrix operand order repaired and checked by reading, not compiled, not run on hardware
//
#include <hip/hip_runtime.h>

#define BATCH 32
#define CH    256
#define HW    4096
#define KGRP  8
#define GN    (KGRP * HW)     // 32768 elements per dynamic group
#define KSPLIT 2              // split-K over HW per batch
#define NSLICE (BATCH * KSPLIT)

typedef __attribute__((ext_vector_type(16))) __bf16 v16bf;
typedef __attribute__((ext_vector_type(8)))  __bf16 v8bf;
typedef __attribute__((ext_vector_type(8)))  float  v8f;

// round-to-nearest-even fp32 -> bf16, packed pair
__device__ __forceinline__ unsigned int pack_bf16_2(float a, float b) {
    unsigned int ua = __float_as_uint(a);
    unsigned int ub = __float_as_uint(b);
    ua = ua + 0x7FFFu + ((ua >> 16) & 1u);
    ub = ub + 0x7FFFu + ((ub >> 16) & 1u);
    return (ua >> 16) | (ub & 0xFFFF0000u);
}

// ---------------------------------------------------------------------------
// Kernel 1: per-(b,c) spatial sum, sum-of-squares, and 1/max(||x||,1e-12)
// ---------------------------------------------------------------------------
__global__ __launch_bounds__(256) void stats_kernel(const float* __restrict__ x,
                                                    float* __restrict__ s,
                                                    float* __restrict__ sq,
                                                    float* __restrict__ invn) {
    const int bc = blockIdx.x;                       // 0..8191
    const float4* xp = (const float4*)(x + (size_t)bc * HW);
    float a1 = 0.0f, a2 = 0.0f;
    for (int i = threadIdx.x; i < HW / 4; i += 256) {
        float4 v = xp[i];
        a1 += v.x + v.y + v.z + v.w;
        a2 += v.x * v.x + v.y * v.y + v.z * v.z + v.w * v.w;
    }
    __shared__ float l1[256];
    __shared__ float l2[256];
    l1[threadIdx.x] = a1;
    l2[threadIdx.x] = a2;
    __syncthreads();
    for (int off = 128; off > 0; off >>= 1) {
        if (threadIdx.x < off) {
            l1[threadIdx.x] += l1[threadIdx.x + off];
            l2[threadIdx.x] += l2[threadIdx.x + off];
        }
        __syncthreads();
    }
    if (threadIdx.x == 0) {
        float S = l1[0], Q = l2[0];
        s[bc]  = S;
        sq[bc] = Q;
        invn[bc] = 1.0f / fmaxf(sqrtf(Q), 1e-12f);
    }
}

// ---------------------------------------------------------------------------
// Kernel 2: per-batch Gram of normalized channels via bf16 WMMA.
// grid = (4 tiles of 128x128, 32 batches, 2 K-halves); block = 256 (8 waves).
// Wave tile 32x64 (waves 4x2) -> 12 ds_load_b128 per 8 WMMAs (B-frag reuse).
// Register-prefetch double buffering over two LDS buffers.
// Normalization (1/||x||) folded into the fp32->bf16 conversion.
// ---------------------------------------------------------------------------
__global__ __launch_bounds__(256) void gram_kernel(const float* __restrict__ x,
                                                   const float* __restrict__ invn,
                                                   float* __restrict__ part) {
    __shared__ __attribute__((aligned(32))) unsigned short lA[2][128 * 32]; // 2 x 8 KB
    __shared__ __attribute__((aligned(32))) unsigned short lB[2][128 * 32]; // 2 x 8 KB

    const int tile = blockIdx.x;           // 0..3
    const int b    = blockIdx.y;           // 0..31
    const int kz   = blockIdx.z;           // 0..1 (split-K)
    const int c0 = (tile >> 1) * 128;
    const int d0 = (tile & 1) * 128;

    const int t     = threadIdx.x;
    const int seg   = t & 7;               // K segment (4 floats)
    const int rbase = t >> 3;              // 0..31
    const int lane  = t & 31;
    const int wave  = t >> 5;              // 0..7
    const int lm    = lane & 15;
    const int kh    = lane >> 4;
    const int wr    = wave >> 1;           // 0..3 : 32-row strip
    const int wc    = wave & 1;            // 0..1 : 64-col strip

    const float* xb  = x + (size_t)b * CH * HW + kz * (HW / KSPLIT);
    const float* inb = invn + b * CH;

    float scA[4], scB[4];
    for (int r = 0; r < 4; ++r) {
        scA[r] = inb[c0 + rbase + 32 * r];
        scB[r] = inb[d0 + rbase + 32 * r];
    }

    v8f acc[2][4] = {};

    // ---- prologue: prefetch K-chunk 0 into registers ----
    float4 va[4], vb[4];
    for (int r = 0; r < 4; ++r) {
        const int row = rbase + 32 * r;
        va[r] = *(const float4*)(xb + (size_t)(c0 + row) * HW + seg * 4);
        vb[r] = *(const float4*)(xb + (size_t)(d0 + row) * HW + seg * 4);
    }

    const int NIT = (HW / KSPLIT) / 32;    // 64 iterations
    int p = 0;
    for (int it = 0; it < NIT; ++it) {
        char* lAp = (char*)lA[p];
        char* lBp = (char*)lB[p];
        // ---- scale, convert, stage prefetched chunk into LDS buffer p ----
        for (int r = 0; r < 4; ++r) {
            const int row = rbase + 32 * r;
            unsigned int a0 = pack_bf16_2(va[r].x * scA[r], va[r].y * scA[r]);
            unsigned int a1 = pack_bf16_2(va[r].z * scA[r], va[r].w * scA[r]);
            unsigned int b0 = pack_bf16_2(vb[r].x * scB[r], vb[r].y * scB[r]);
            unsigned int b1 = pack_bf16_2(vb[r].z * scB[r], vb[r].w * scB[r]);
            *(uint2*)(lAp + row * 64 + seg * 8) = make_uint2(a0, a1);
            *(uint2*)(lBp + row * 64 + seg * 8) = make_uint2(b0, b1);
        }
        __syncthreads();   // emits s_wait_dscnt 0 + barrier: also retires prior frag reads

        // ---- prefetch next K-chunk (overlaps with WMMA below) ----
        if (it + 1 < NIT) {
            const int kb = (it + 1) * 32;
            for (int r = 0; r < 4; ++r) {
                const int row = rbase + 32 * r;
                va[r] = *(const float4*)(xb + (size_t)(c0 + row) * HW + kb + seg * 4);
                vb[r] = *(const float4*)(xb + (size_t)(d0 + row) * HW + kb + seg * 4);
            }
        }

        // ---- WMMA: wave computes 32x64 = 2x4 tiles of 16x16 ----
        union { v8bf h[2]; v16bf v; } fa[2];
#pragma unroll
        for (int i = 0; i < 2; ++i) {
            const int M = wr * 32 + i * 16 + lm;
            fa[i].h[0] = *(const v8bf*)(lAp + M * 64 + kh * 32);       // K = 16*kh+0..7
            fa[i].h[1] = *(const v8bf*)(lAp + M * 64 + kh * 32 + 16);  // K = 16*kh+8..15
        }
#pragma unroll
        for (int j = 0; j < 4; ++j) {
            const int n = wc * 64 + 16 * j + lm;
            v16bf fb = *(const v16bf*)(lBp + n * 64 + kh * 32);        // K = 16*kh+0..15
            acc[0][j] = __builtin_amdgcn_wmma_f32_16x16x32_bf16(
                false, fa[0].v, false, fb, (short)0, acc[0][j], false, false);
            acc[1][j] = __builtin_amdgcn_wmma_f32_16x16x32_bf16(
                false, fa[1].v, false, fb, (short)0, acc[1][j], false, false);
        }
        p ^= 1;
    }

    // ---- write partial Gram slice (deterministic split-K) ----
    float* pb = part + ((size_t)(b * KSPLIT + kz) << 16);
#pragma unroll
    for (int i = 0; i < 2; ++i) {
#pragma unroll
        for (int j = 0; j < 4; ++j) {
            const int n = d0 + wc * 64 + 16 * j + lm;
#pragma unroll
            for (int v = 0; v < 8; ++v) {
                const int m = c0 + wr * 32 + i * 16 + v + 8 * kh;
                pb[m * 256 + n] = acc[i][j][v];
            }
        }
    }
}

// ---------------------------------------------------------------------------
// Kernel 3: reduce partial slices, then select 8 smallest |cos| per row
// (tie-break: lowest index first, matching jax.lax.top_k).
// ---------------------------------------------------------------------------
__global__ __launch_bounds__(256) void topk_kernel(const float* __restrict__ part,
                                                   int* __restrict__ idxo) {
    const int c = blockIdx.x;
    const int d = threadIdx.x;
    float acc = 0.0f;
    for (int sl = 0; sl < NSLICE; ++sl)
        acc += part[((size_t)sl << 16) + c * 256 + d];
    float myv = fabsf(acc * (1.0f / (float)BATCH));

    __shared__ float sv[256];
    __shared__ int   si[256];
    for (int r = 0; r < KGRP; ++r) {
        sv[d] = myv;
        si[d] = d;
        __syncthreads();
        for (int off = 128; off > 0; off >>= 1) {
            if (d < off) {
                float v2 = sv[d + off]; int i2 = si[d + off];
                float v1 = sv[d];       int i1 = si[d];
                if (v2 < v1 || (v2 == v1 && i2 < i1)) { sv[d] = v2; si[d] = i2; }
            }
            __syncthreads();
        }
        const int w = si[0];
        if (d == 0) idxo[c * KGRP + r] = w;
        if (d == w) myv = 3.0e38f;   // remove winner for next round
        __syncthreads();
    }
}

// ---------------------------------------------------------------------------
// Kernel 4: dynamic-group statistics -> mean, inv-std per (b,c)
// ---------------------------------------------------------------------------
__global__ __launch_bounds__(256) void groupstats_kernel(const float* __restrict__ s,
                                                         const float* __restrict__ sq,
                                                         const int* __restrict__ idxo,
                                                         float* __restrict__ meanv,
                                                         float* __restrict__ istd) {
    const int b = blockIdx.x;
    const int c = threadIdx.x;
    const float* sb = s + b * CH;
    const float* qb = sq + b * CH;
    float gs = sb[c], gq = qb[c];
#pragma unroll
    for (int i = 1; i < KGRP; ++i) {          // skip idx[:,0] per reference
        const int dd = idxo[c * KGRP + i];
        gs += sb[dd];
        gq += qb[dd];
    }
    const float mean = gs * (1.0f / (float)GN);
    const float var  = gq * (1.0f / (float)GN) - mean * mean;
    meanv[b * CH + c] = mean;
    istd[b * CH + c]  = rsqrtf(var + 1e-5f);
}

// ---------------------------------------------------------------------------
// Kernel 5: out = gamma*(x-mean)*invstd + beta, float4-vectorized
// ---------------------------------------------------------------------------
__global__ __launch_bounds__(256) void apply_kernel(const float* __restrict__ x,
                                                    const float* __restrict__ gamma,
                                                    const float* __restrict__ beta,
                                                    const float* __restrict__ meanv,
                                                    const float* __restrict__ istd,
                                                    float* __restrict__ out) {
    const size_t gid = (size_t)blockIdx.x * 256 + threadIdx.x;  // float4 index
    const int bc = (int)(gid >> 10);        // HW/4 = 1024 float4 per (b,c)
    const int c  = bc & (CH - 1);
    const float sc = gamma[c] * istd[bc];
    const float bi = beta[c] - sc * meanv[bc];
    float4 v = ((const float4*)x)[gid];
    v.x = v.x * sc + bi;
    v.y = v.y * sc + bi;
    v.z = v.z * sc + bi;
    v.w = v.w * sc + bi;
    ((float4*)out)[gid] = v;
}

// ---------------------------------------------------------------------------
extern "C" void kernel_launch(void* const* d_in, const int* in_sizes, int n_in,
                              void* d_out, int out_size, void* d_ws, size_t ws_size,
                              hipStream_t stream) {
    const float* x     = (const float*)d_in[0];
    const float* gamma = (const float*)d_in[1];
    const float* beta  = (const float*)d_in[2];
    float* out = (float*)d_out;

    // workspace layout (floats): ~16.9 MB
    float* ws    = (float*)d_ws;
    float* s     = ws;                         // 8192
    float* sq    = s + BATCH * CH;             // 8192
    float* invn  = sq + BATCH * CH;            // 8192
    float* meanv = invn + BATCH * CH;          // 8192
    float* istd  = meanv + BATCH * CH;         // 8192
    int*   idx   = (int*)(istd + BATCH * CH);  // 256*8 ints
    float* part  = (float*)(idx + CH * KGRP);  // 64*256*256 floats

    stats_kernel<<<BATCH * CH, 256, 0, stream>>>(x, s, sq, invn);
    gram_kernel<<<dim3(4, BATCH, KSPLIT), 256, 0, stream>>>(x, invn, part);
    topk_kernel<<<CH, 256, 0, stream>>>(part, idx);
    groupstats_kernel<<<BATCH, 256, 0, stream>>>(s, sq, idx, meanv, istd);
    apply_kernel<<<(BATCH * CH * HW) / 4 / 256, 256, 0, stream>>>(x, gamma, beta, meanv, istd, out);
}
